// PointCNN_32650341384580
// MI455X (gfx1250) — matrix operand order repaired and checked by reading, not compiled, not run on hardware
//
#include <hip/hip_runtime.h>
#include <hip/hip_bf16.h>

// ---------------------------------------------------------------------------
// PointCNN forward for MI455X (gfx1250, wave32, WMMA).
// Heavy GEMMs use v_wmma_f32_16x16x32_f16 (f16 in, f32 accum); LDS tiles are
// loaded with async global->LDS b128 transfers (ASYNCcnt-tracked) and
// fragments are assembled as contiguous 16B/32B LDS vector loads per the
// documented wave32 VGPR layouts.
// ---------------------------------------------------------------------------

typedef __attribute__((ext_vector_type(16))) _Float16 v16h;
typedef __attribute__((ext_vector_type(8)))  _Float16 v8h;
typedef __attribute__((ext_vector_type(8)))  float    v8f;
typedef __attribute__((ext_vector_type(4)))  int      v4i;
typedef unsigned int u32;

#define TM 32
#define TN 64
#define TK 32

#if defined(__has_builtin)
#if __has_builtin(__builtin_amdgcn_global_load_async_to_lds_b128) && \
    __has_builtin(__builtin_amdgcn_s_wait_asynccnt)
#define HAVE_ASYNC_LDS 1
#endif
#endif
#ifndef HAVE_ASYNC_LDS
#define HAVE_ASYNC_LDS 0
#endif

#if HAVE_ASYNC_LDS
typedef __attribute__((address_space(1))) v4i gv4i;   // global int4
typedef __attribute__((address_space(3))) v4i lv4i;   // LDS int4
#endif

// -------------------------- WMMA GEMM -------------------------------------
// C[R,N] = act( A[R,Kp](f16) @ Bt[Np,Kp](f16, pre-transposed) + bias[N] )
// Requirements: R % 32 == 0, Kp % 32 == 0, Np % 64 == 0 (operands zero-padded).
__global__ __launch_bounds__(256)
void wmma_gemm_bias_act(const _Float16* __restrict__ A,
                        const _Float16* __restrict__ Bt,
                        const float* __restrict__ bias,
                        float* __restrict__ C,
                        int R, int Kp, int N, int Np, int act) {
  __shared__ __align__(64) u32 sAu[TM * TK / 2];   // [32 rows][32 halves]
  __shared__ __align__(64) u32 sBu[TN * TK / 2];   // [64 n-rows][32 halves] (K-contig)

  const int tilesN = Np / TN;
  const int bm = blockIdx.x / tilesN;
  const int bn = blockIdx.x % tilesN;
  const int tid  = threadIdx.x;
  const int lane = tid & 31;
  const int wave = tid >> 5;       // 0..7
  const int wr   = wave & 1;       // 16-row tile within 32
  const int wc   = wave >> 1;      // 16-col tile within 64
  const int grp  = lane >> 4;      // lane group (0: lanes 0-15, 1: 16-31)
  const int l16  = lane & 15;

  v8f acc = {0.f, 0.f, 0.f, 0.f, 0.f, 0.f, 0.f, 0.f};

  const size_t rowBase = (size_t)bm * TM;

  for (int kt = 0; kt < Kp; kt += TK) {
#if HAVE_ASYNC_LDS
    {
      // Async global->LDS tile copies (ASYNCcnt-tracked, 16B per lane).
      const char* gA = (const char*)(A + rowBase * Kp + kt);
      const char* gB = (const char*)(Bt + ((size_t)bn * TN) * Kp + kt);
      char* lA = (char*)sAu;
      char* lB = (char*)sBu;
      int r = tid >> 2, c4 = (tid & 3) * 16;
      if (tid < 128) {   // A: 32 rows x 64B
        __builtin_amdgcn_global_load_async_to_lds_b128(
            (gv4i*)(gA + (size_t)r * (Kp * 2) + c4),
            (lv4i*)(lA + r * 64 + c4), 0, 0);
      }
      {                  // B: 64 rows x 64B
        __builtin_amdgcn_global_load_async_to_lds_b128(
            (gv4i*)(gB + (size_t)r * (Kp * 2) + c4),
            (lv4i*)(lB + r * 64 + c4), 0, 0);
      }
      __builtin_amdgcn_s_wait_asynccnt(0);
    }
    __syncthreads();
#else
    {
      const int kd = Kp >> 1;                       // dwords per K row
      const u32* Ag = (const u32*)(A + rowBase * Kp + kt);
      #pragma unroll
      for (int i = 0; i < 2; ++i) {
        int t = tid + i * 256;                      // 512 dwords
        int r = t >> 4, c = t & 15;
        sAu[r * 16 + c] = Ag[(size_t)r * kd + c];
      }
      const u32* Bg = (const u32*)(Bt + ((size_t)bn * TN) * Kp + kt);
      #pragma unroll
      for (int i = 0; i < 4; ++i) {
        int t = tid + i * 256;                      // 1024 dwords
        int r = t >> 4, c = t & 15;
        sBu[r * 16 + c] = Bg[(size_t)r * kd + c];
      }
    }
    __syncthreads();
#endif

    // prefetch next K tile into cache (global_prefetch_b8)
    if (kt + TK < Kp) {
      __builtin_prefetch(A + (rowBase + lane) * Kp + kt + TK, 0, 3);
      __builtin_prefetch(Bt + ((size_t)bn * TN + lane) * Kp + kt + TK, 0, 3);
    }

    // --- fragments: contiguous LDS vector loads per wave32 layouts ---
    const _Float16* sA  = (const _Float16*)sAu;
    const _Float16* sBt = (const _Float16*)sBu;
    // A 16x32 f16: lane holds K = {8g..8g+7} U {16+8g..16+8g+7} of its row.
    const _Float16* sAr = sA + (wr * 16 + l16) * TK;
    v8h a0 = *(const v8h*)(sAr + grp * 8);
    v8h a1 = *(const v8h*)(sAr + 16 + grp * 8);
    v16h a = __builtin_shufflevector(a0, a1, 0, 1, 2, 3, 4, 5, 6, 7,
                                     8, 9, 10, 11, 12, 13, 14, 15);
    // B 32x16 f16: lane holds K = {8g..8g+7} U {16+8g..16+8g+7} of its column (K-contig in sBt).
    const _Float16* sBr = sBt + (wc * 16 + l16) * TK;
    v8h b0 = *(const v8h*)(sBr + grp * 8);
    v8h b1 = *(const v8h*)(sBr + 16 + grp * 8);
    v16h b = __builtin_shufflevector(b0, b1, 0, 1, 2, 3, 4, 5, 6, 7,
                                     8, 9, 10, 11, 12, 13, 14, 15);

    acc = __builtin_amdgcn_wmma_f32_16x16x32_f16(
        /*neg_a=*/false, a, /*neg_b=*/false, b,
        /*c_mod=*/(short)0, acc, /*reuse_a=*/false, /*reuse_b=*/false);
    __syncthreads();
  }

  // --- store: C/D layout: VGPR r -> M = r + 8*grp, N = lane%16 ---
  int col = bn * TN + wc * 16 + l16;
  if (col < N) {
    float bv = bias ? bias[col] : 0.f;
    #pragma unroll
    for (int r = 0; r < 8; ++r) {
      int row = (int)rowBase + wr * 16 + r + 8 * grp;
      if (row < R) {
        float v = acc[r] + bv;
        if (act) v = v > 0.f ? v : (__expf(v) - 1.f);   // ELU(alpha=1)
        C[(size_t)row * N + col] = v;
      }
    }
  }
}

// ----------------------- Farthest point sampling ---------------------------
__global__ __launch_bounds__(256)
void fps_kernel(const float* __restrict__ p, int N, int m, int* __restrict__ idx) {
  __shared__ float sdist[8192];
  __shared__ float rv[256];
  __shared__ int   ri[256];
  __shared__ float lp[3];
  __shared__ int   lastIdx;
  const int b = blockIdx.x, tid = threadIdx.x;
  const float* pb = p + (size_t)b * N * 3;

  for (int i = tid; i < N; i += 256) sdist[i] = 1e10f;
  if (tid == 0) { lastIdx = 0; idx[(size_t)b * m] = 0; }
  __syncthreads();

  for (int it = 1; it < m; ++it) {
    if (tid < 3) lp[tid] = pb[(size_t)lastIdx * 3 + tid];
    __syncthreads();
    float lx = lp[0], ly = lp[1], lz = lp[2];
    float bv = -1.f; int bi = 0;
    for (int i = tid; i < N; i += 256) {
      float dx = pb[i * 3 + 0] - lx;
      float dy = pb[i * 3 + 1] - ly;
      float dz = pb[i * 3 + 2] - lz;
      float d  = dx * dx + dy * dy + dz * dz;
      float nd = fminf(sdist[i], d);
      sdist[i] = nd;
      if (nd > bv) { bv = nd; bi = i; }
    }
    rv[tid] = bv; ri[tid] = bi;
    __syncthreads();
    for (int s = 128; s > 0; s >>= 1) {
      if (tid < s) {
        if (rv[tid + s] > rv[tid] ||
            (rv[tid + s] == rv[tid] && ri[tid + s] < ri[tid])) {
          rv[tid] = rv[tid + s]; ri[tid] = ri[tid + s];
        }
      }
      __syncthreads();
    }
    if (tid == 0) { lastIdx = ri[0]; idx[(size_t)b * m + it] = ri[0]; }
    __syncthreads();
  }
}

// q[b,i,:] = p[b, idx[b,i], :]
__global__ void gather3_kernel(const float* __restrict__ p, const int* __restrict__ idx,
                               float* __restrict__ q, int M, int N, int total) {
  int t = blockIdx.x * blockDim.x + threadIdx.x;
  if (t >= total) return;
  int b = t / M;
  int j = idx[t];
  const float* s = p + ((size_t)b * N + j) * 3;
  q[(size_t)t * 3 + 0] = s[0];
  q[(size_t)t * 3 + 1] = s[1];
  q[(size_t)t * 3 + 2] = s[2];
}

// ------------------------------- kNN ---------------------------------------
__global__ __launch_bounds__(32)
void knn_kernel(const float* __restrict__ q, const float* __restrict__ p,
                int M, int N, int K, int dil, int k, int* __restrict__ out) {
  __shared__ float sd[8192];
  const int bm = blockIdx.x;
  const int b = bm / M;
  const int lane = threadIdx.x;
  const float* pb = p + (size_t)b * N * 3;
  const float qx = q[(size_t)bm * 3 + 0];
  const float qy = q[(size_t)bm * 3 + 1];
  const float qz = q[(size_t)bm * 3 + 2];
  for (int j = lane; j < N; j += 32) {
    float dx = pb[j * 3 + 0] - qx;
    float dy = pb[j * 3 + 1] - qy;
    float dz = pb[j * 3 + 2] - qz;
    sd[j] = dx * dx + dy * dy + dz * dz;
  }
  __syncthreads();
  for (int sel = 0; sel < K; ++sel) {
    float bv = 3.0e38f; int bi = 0x7fffffff;
    for (int j = lane; j < N; j += 32) {
      float d = sd[j];
      if (d < bv || (d == bv && j < bi)) { bv = d; bi = j; }
    }
    #pragma unroll
    for (int off = 16; off > 0; off >>= 1) {
      float ov = __shfl_down(bv, off);
      int   oi = __shfl_down(bi, off);
      if (ov < bv || (ov == bv && oi < bi)) { bv = ov; bi = oi; }
    }
    bi = __shfl(bi, 0);
    if (lane == 0) {
      if (sel % dil == 0) out[(size_t)bm * k + sel / dil] = bi;
      sd[bi] = 3.4e38f;
    }
    __syncthreads();
  }
}

// d[b,m,kk,:] = p[b, idx[b,m,kk], :] - q[b,m,:]
__global__ void build_d_kernel(const float* __restrict__ p, const float* __restrict__ q,
                               const int* __restrict__ idx, float* __restrict__ d,
                               int M, int N, int k, int total) {
  int t = blockIdx.x * blockDim.x + threadIdx.x;
  if (t >= total) return;                 // total = B*M*k
  int bm = t / k;                         // b*M + m
  int b  = bm / M;
  int j  = idx[t];
  const float* ps = p + ((size_t)b * N + j) * 3;
  const float* qs = q + (size_t)bm * 3;
  d[(size_t)t * 3 + 0] = ps[0] - qs[0];
  d[(size_t)t * 3 + 1] = ps[1] - qs[1];
  d[(size_t)t * 3 + 2] = ps[2] - qs[2];
}

// feats16[row, 0:cmid] = f[row,:];  feats16[row, cmid:cmid+cin] = x[b, idx[row], :]
__global__ void concat_feats_kernel(const float* __restrict__ f, const float* __restrict__ x,
                                    const int* __restrict__ idx, _Float16* __restrict__ feats,
                                    int M, int N, int k, int cmid, int cin, size_t total) {
  const int Cc = cmid + cin;
  for (size_t t = (size_t)blockIdx.x * blockDim.x + threadIdx.x; t < total;
       t += (size_t)gridDim.x * blockDim.x) {
    size_t row = t / Cc;
    int c = (int)(t % Cc);
    float v;
    if (c < cmid) {
      v = f[row * cmid + c];
    } else {
      int b = (int)(row / ((size_t)M * k));
      int j = idx[row];
      v = x[((size_t)b * N + j) * cin + (c - cmid)];
    }
    feats[t] = (_Float16)v;
  }
}

// Per-point X-transform: fx[b,m,kk,c] = sum_j X[b,m,kk,j] * feats[b,m,j,c]
__global__ __launch_bounds__(256)
void xapply_kernel(const float* __restrict__ X, const _Float16* __restrict__ feats,
                   _Float16* __restrict__ fx, int k, int C, int Kp) {
  __shared__ float sX[256];              // k*k <= 256
  const int bm = blockIdx.x;
  const int tid = threadIdx.x;
  const int kk2 = k * k;
  if (tid < kk2) sX[tid] = X[(size_t)bm * kk2 + tid];
  __syncthreads();
  const _Float16* fr = feats + (size_t)bm * k * C;
  _Float16* fo = fx + (size_t)bm * Kp;
  for (int t = tid; t < k * C; t += 256) {
    int kk = t / C, c = t % C;
    float acc = 0.f;
    for (int j = 0; j < k; ++j) acc += sX[kk * k + j] * (float)fr[j * C + c];
    fo[kk * C + c] = (_Float16)acc;
  }
}

// f32 [r,c] -> f16 [rp,cp] with zero padding (row-major, for GEMM A operand)
__global__ void convert_pad2_kernel(const float* __restrict__ src, _Float16* __restrict__ dst,
                                    int r, int c, int rp, int cp) {
  size_t total = (size_t)rp * cp;
  for (size_t t = (size_t)blockIdx.x * blockDim.x + threadIdx.x; t < total;
       t += (size_t)gridDim.x * blockDim.x) {
    int i = (int)(t / cp), j = (int)(t % cp);
    float v = (i < r && j < c) ? src[(size_t)i * c + j] : 0.0f;
    dst[t] = (_Float16)v;
  }
}

// f32 [r,c] -> f16 TRANSPOSED [cp, rp] with zero padding (GEMM B operand, K-contig)
__global__ void convert_pad2_t_kernel(const float* __restrict__ src, _Float16* __restrict__ dst,
                                      int r, int c, int rp, int cp) {
  size_t total = (size_t)cp * rp;
  for (size_t t = (size_t)blockIdx.x * blockDim.x + threadIdx.x; t < total;
       t += (size_t)gridDim.x * blockDim.x) {
    int j = (int)(t / rp), i = (int)(t % rp);   // dst[j, i] = src[i, j]
    float v = (i < r && j < c) ? src[(size_t)i * c + j] : 0.0f;
    dst[t] = (_Float16)v;
  }
}

__global__ void fill_zero_f16(_Float16* __restrict__ p, size_t n) {
  for (size_t t = (size_t)blockIdx.x * blockDim.x + threadIdx.x; t < n;
       t += (size_t)gridDim.x * blockDim.x) p[t] = (_Float16)0.0f;
}

// ---------------------------------------------------------------------------
// Host orchestration
// ---------------------------------------------------------------------------
static inline size_t alignup(size_t x, size_t a) { return (x + a - 1) & ~(a - 1); }
static inline size_t cdiv(size_t a, size_t b) { return (a + b - 1) / b; }
static inline size_t smin(size_t a, size_t b) { return a < b ? a : b; }
static inline int pad32(int x) { return (x + 31) & ~31; }
static inline int pad64(int x) { return (x + 63) & ~63; }

extern "C" void kernel_launch(void* const* d_in, const int* in_sizes, int n_in,
                              void* d_out, int out_size, void* d_ws, size_t ws_size,
                              hipStream_t stream) {
  (void)in_sizes; (void)n_in; (void)out_size; (void)ws_size;

  const float* P  = (const float*)d_in[0];   // [4, 8192, 3]
  const float* X0 = (const float*)d_in[1];   // [4, 8192, 3]
  auto PRM = [&](int l, int j) -> const float* {
    return (const float*)d_in[2 + 12 * l + j];
  }; // 0:W1 1:b1 2:W2 3:b2 4:Wx1 5:bx1 6:Wx2 7:bx2 8:Wx3 9:bx3 10:Wc 11:bc

  static const int CIN_[7]  = {3, 256, 256, 512, 1024, 512, 256};
  static const int COUT_[7] = {256, 256, 512, 1024, 512, 256, 256};
  static const int KK_[7]   = {8, 12, 16, 16, 16, 12, 8};
  static const int DIL_[7]  = {1, 2, 2, 4, 6, 4, 4};
  static const int NSRC_[7] = {8192, 2048, 768, 384, 128, 384, 768};
  static const int MQ_[7]   = {2048, 768, 384, 128, 384, 768, 2048};

  // ---- workspace layout (bump allocator over d_ws) ----
  char* ws = (char*)d_ws;
  size_t off = 0;
  auto alloc = [&](size_t bytes) -> void* {
    void* ptr = ws + off;
    off = alignup(off + bytes, 256);
    return ptr;
  };
  // persistent
  float* pl1 = (float*)alloc((size_t)4 * 2048 * 3 * 4);
  float* pl2 = (float*)alloc((size_t)4 * 768  * 3 * 4);
  float* pl3 = (float*)alloc((size_t)4 * 384  * 3 * 4);
  float* pl4 = (float*)alloc((size_t)4 * 128  * 3 * 4);
  float* xb0 = (float*)alloc((size_t)4 * 2048 * 256 * 4);
  float* xb1 = (float*)alloc((size_t)4 * 2048 * 256 * 4);
  // per-layer scratch (max sizes across all layers)
  int*       fidx    = (int*)alloc((size_t)4 * 2048 * 4);
  int*       knnidx  = (int*)alloc((size_t)65536 * 4);
  float*     dflat   = (float*)alloc((size_t)196608 * 4);
  _Float16*  Ah      = (_Float16*)alloc((size_t)4194304 * 2);
  _Float16*  Bh      = (_Float16*)alloc((size_t)12582912 * 2);
  float*     f_a     = (float*)alloc((size_t)4194304 * 4);
  float*     f_b     = (float*)alloc((size_t)4194304 * 4);
  float*     X_a     = (float*)alloc((size_t)524288 * 4);
  float*     X_b     = (float*)alloc((size_t)524288 * 4);
  _Float16*  feats16 = (_Float16*)alloc((size_t)28311552 * 2);
  _Float16*  fx16    = (_Float16*)alloc((size_t)28311552 * 2);

  const float* srcP[7] = {P, pl1, pl2, pl3, pl4, pl3, pl2};
  float*       qP[7]   = {pl1, pl2, pl3, pl4, pl3, pl2, pl1};

  auto cvt = [&](const float* src, _Float16* dst, int r, int c, int rp, int cp) {
    size_t tot = (size_t)rp * cp;
    int blocks = (int)smin((size_t)4096, cdiv(tot, 256));
    convert_pad2_kernel<<<blocks, 256, 0, stream>>>(src, dst, r, c, rp, cp);
  };
  // weights: [K,N] f32 -> [Np,Kp] f16 (transposed, K-contiguous)
  auto cvtT = [&](const float* src, _Float16* dst, int r, int c, int rp, int cp) {
    size_t tot = (size_t)rp * cp;
    int blocks = (int)smin((size_t)4096, cdiv(tot, 256));
    convert_pad2_t_kernel<<<blocks, 256, 0, stream>>>(src, dst, r, c, rp, cp);
  };
  auto gemm = [&](const _Float16* A, const _Float16* B, const float* bias, float* C,
                  int R, int Kp, int N, int Np, int act) {
    dim3 g((unsigned)((R / TM) * (Np / TN)));
    wmma_gemm_bias_act<<<g, 256, 0, stream>>>(A, B, bias, C, R, Kp, N, Np, act);
  };

  const float* xin = X0;
  for (int l = 0; l < 7; ++l) {
    const int cin = CIN_[l], cout = COUT_[l], k = KK_[l], dil = DIL_[l];
    const int N = NSRC_[l], M = MQ_[l];
    const int cmid = cout / 4, C = cmid + cin, Kn = k * dil;
    const int Rk = 4 * M * k, Rm = 4 * M;
    const float* psrc = srcP[l];
    float* q = qP[l];

    // --- FPS downsampling (encoder layers only) ---
    if (l < 4) {
      fps_kernel<<<4, 256, 0, stream>>>(psrc, N, M, fidx);
      int tot = 4 * M;
      gather3_kernel<<<(tot + 255) / 256, 256, 0, stream>>>(psrc, fidx, q, M, N, tot);
    }

    // --- dilated kNN ---
    knn_kernel<<<4 * M, 32, 0, stream>>>(q, psrc, M, N, Kn, dil, k, knnidx);

    // --- local coords d (also serves as flattened [B*M, 3k]) ---
    build_d_kernel<<<(Rk + 255) / 256, 256, 0, stream>>>(psrc, q, knnidx, dflat, M, N, k, Rk);

    // --- MLP_delta: f = elu(elu(d@W1+b1)@W2+b2) ---
    cvt(dflat, Ah, Rk, 3, Rk, 32);
    cvtT(PRM(l, 0), Bh, 3, cmid, 32, pad64(cmid));
    gemm(Ah, Bh, PRM(l, 1), f_a, Rk, 32, cmid, pad64(cmid), 1);
    cvt(f_a, Ah, Rk, cmid, Rk, cmid);
    cvtT(PRM(l, 2), Bh, cmid, cmid, cmid, pad64(cmid));
    gemm(Ah, Bh, PRM(l, 3), f_b, Rk, cmid, cmid, pad64(cmid), 1);

    // --- feats = concat(f, gather(x, idx)) in f16 ---
    {
      size_t tot = (size_t)Rk * C;
      concat_feats_kernel<<<4096, 256, 0, stream>>>(f_b, xin, knnidx, feats16,
                                                    M, N, k, cmid, cin, tot);
    }

    // --- X-transformation: 3 dense layers on [B*M, 3k] ---
    const int k3 = 3 * k, kk2 = k * k;
    cvt(dflat, Ah, Rm, k3, Rm, pad32(k3));
    cvtT(PRM(l, 4), Bh, k3, kk2, pad32(k3), pad64(kk2));
    gemm(Ah, Bh, PRM(l, 5), X_a, Rm, pad32(k3), kk2, pad64(kk2), 1);
    cvt(X_a, Ah, Rm, kk2, Rm, pad32(kk2));
    cvtT(PRM(l, 6), Bh, kk2, kk2, pad32(kk2), pad64(kk2));
    gemm(Ah, Bh, PRM(l, 7), X_b, Rm, pad32(kk2), kk2, pad64(kk2), 1);
    cvt(X_b, Ah, Rm, kk2, Rm, pad32(kk2));
    cvtT(PRM(l, 8), Bh, kk2, kk2, pad32(kk2), pad64(kk2));
    gemm(Ah, Bh, PRM(l, 9), X_a, Rm, pad32(kk2), kk2, pad64(kk2), 0);

    // --- per-point X @ feats -> fx (f16, padded for GEMM A) ---
    const int KC = k * C, KCp = pad32(KC);
    {
      size_t tot = (size_t)Rm * KCp;
      fill_zero_f16<<<4096, 256, 0, stream>>>(fx16, tot);
    }
    xapply_kernel<<<Rm, 256, 0, stream>>>(X_a, feats16, fx16, k, C, KCp);

    // --- final projection: out[b,m,o] = fx.flat @ Wc.flat + bc ---
    cvtT(PRM(l, 10), Bh, KC, cout, KCp, pad64(cout));
    float* xout = (l == 6) ? (float*)d_out : ((l % 2 == 0) ? xb0 : xb1);
    gemm(fx16, Bh, PRM(l, 11), xout, Rm, KCp, cout, pad64(cout), 0);

    xin = xout;
  }
}
